// PANPooling_56908316672611
// MI455X (gfx1250) — matrix operand order repaired and checked by reading, not compiled, not run on hardware
//
#include <hip/hip_runtime.h>
#include <hip/hip_bf16.h>

// ---------------- problem constants (fixed by reference) ----------------
#define NNODES   100000
#define NEDGES   3200000
#define DIM      256
#define KSEL     50000          // ceil(0.5 * NNODES)
#define TILE     2048           // radix sort items per block
#define RADIX    256

// output section offsets (float elements, concatenated in return order)
#define O_XNEW   0                       // 50000*256 = 12,800,000
#define O_EI     12800000                // 2*3,200,000
#define O_EA     19200000                // 3,200,000
#define O_BATCH  22400000                // 50,000
#define O_PERM   22450000                // 50,000
#define O_MASK   22500000                // 3,200,000

typedef __attribute__((ext_vector_type(16))) _Float16 v16h;
typedef __attribute__((ext_vector_type(8)))  float    v8f;
typedef unsigned long long u64;
typedef unsigned int       u32;

static __device__ __forceinline__ int imin(int a, int b) { return a < b ? a : b; }

// ---------------------------------------------------------------
// 1) score1 = x @ p  via v_wmma_f32_16x16x32_f16 (16-row tile per wave)
// ---------------------------------------------------------------
__global__ void __launch_bounds__(256)
gemv_wmma(const float* __restrict__ x, const float* __restrict__ p,
          float* __restrict__ score1, int nTiles) {
  int wave = blockIdx.x * (blockDim.x >> 5) + (threadIdx.x >> 5);
  if (wave >= nTiles) return;              // wave-uniform: EXEC stays all-ones
  int lane = threadIdx.x & 31;
  int m    = lane & 15;                    // row within tile / N column
  int hi   = lane >> 4;                    // lane group 0/1
  int r0   = wave << 4;
  const float* xr = x + ((size_t)(r0 + m) << 8);   // row of x, stride DIM=256

  v8f acc = {};
#pragma unroll
  for (int k0 = 0; k0 < DIM; k0 += 32) {
    int kb = k0 + hi * 8;
    __builtin_prefetch(xr + kb + 64, 0, 0);
    // A fragment: 16-bit A 16x32 layout (ISA 7.12.2):
    //   comps 0..7  -> K = kb + c ; comps 8..15 -> K = kb + c + 8
    v16h a, b;
#pragma unroll
    for (int c = 0; c < 8; ++c)  a[c] = (_Float16)xr[kb + c];
#pragma unroll
    for (int c = 8; c < 16; ++c) a[c] = (_Float16)xr[kb + c + 8];
    // B fragment: 32x16, comp c -> K = 8*hi + c (c<8), 8*hi + c + 8 (c>=8), identical across N columns
    const float* pb = p + kb;
#pragma unroll
    for (int c = 0; c < 16; ++c) b[c] = (_Float16)pb[c < 8 ? c : c + 8];
    acc = __builtin_amdgcn_wmma_f32_16x16x32_f16(false, a, false, b,
                                                 (short)0, acc, false, false);
  }
  // D layout: acc[r]: lanes 0-15 -> M=r, lanes 16-31 -> M=r+8; every N identical
  if (m == 0) {
    int base = r0 + hi * 8;
#pragma unroll
    for (int r = 0; r < 8; ++r) score1[base + r] = acc[r];
  }
}

// ---------------------------------------------------------------
// 2) misc elementwise kernels
// ---------------------------------------------------------------
__global__ void fill_u32(u32* ptr, u32 val, int n) {
  int i = blockIdx.x * blockDim.x + threadIdx.x;
  if (i < n) ptr[i] = val;
}

__global__ void seg_sum(const int* __restrict__ col, const float* __restrict__ w,
                        float* score2, int ne) {
  int e = blockIdx.x * blockDim.x + threadIdx.x;
  if (e < ne) atomicAdd(&score2[col[e]], w[e]);
}

__global__ void score_kernel(const float* __restrict__ s1, const float* __restrict__ s2,
                             const float* __restrict__ beta, float* score, int n) {
  int i = blockIdx.x * blockDim.x + threadIdx.x;
  if (i < n) score[i] = tanhf(beta[0] * s1[i] + beta[1] * s2[i]);
}

// descending-score key with index tiebreak (matches lax.top_k ordering)
__global__ void build_node_keys(const float* __restrict__ score, u64* keys, int n) {
  int i = blockIdx.x * blockDim.x + threadIdx.x;
  if (i >= n) return;
  u32 u  = __float_as_uint(score[i]);
  u32 ka = (u & 0x80000000u) ? ~u : (u | 0x80000000u);   // ascending-sortable
  u32 kd = ~ka;                                          // descending
  keys[i] = ((u64)kd << 17) | (u32)i;
}

// stable lexsort((col,row)) key: primary row, secondary col, tiebreak edge id
__global__ void build_edge_keys(const int* __restrict__ ei, u64* keys, int ne) {
  int e = blockIdx.x * blockDim.x + threadIdx.x;
  if (e >= ne) return;
  u64 r = (u32)ei[e];
  u64 c = (u32)ei[ne + e];
  keys[e] = (r << 39) | (c << 22) | (u32)e;
}

// ---------------------------------------------------------------
// 3) stable LSD radix sort (8-bit digits) on u64 keys
// ---------------------------------------------------------------
__global__ void __launch_bounds__(256)
radix_hist(const u64* __restrict__ keys, int n, int shift,
           u32* counts, int numBlocks) {
  __shared__ u32 sh[RADIX];
  int b = blockIdx.x;
  for (int i = threadIdx.x; i < RADIX; i += blockDim.x) sh[i] = 0;
  __syncthreads();
  int start = b * TILE, end = imin(n, start + TILE);
  for (int i = start + threadIdx.x; i < end; i += blockDim.x) {
    u32 d = (u32)((keys[i] >> shift) & 0xFF);
    atomicAdd(&sh[d], 1u);
  }
  __syncthreads();
  for (int i = threadIdx.x; i < RADIX; i += blockDim.x)
    counts[(size_t)i * numBlocks + b] = sh[i];        // digit-major
}

__global__ void __launch_bounds__(256)
radix_digit_totals(const u32* __restrict__ counts, int numBlocks, u32* totals) {
  __shared__ u32 sh[256];
  int d = blockIdx.x;
  u32 s = 0;
  for (int b = threadIdx.x; b < numBlocks; b += blockDim.x)
    s += counts[(size_t)d * numBlocks + b];
  sh[threadIdx.x] = s;
  __syncthreads();
  for (int off = 128; off > 0; off >>= 1) {
    if ((int)threadIdx.x < off) sh[threadIdx.x] += sh[threadIdx.x + off];
    __syncthreads();
  }
  if (threadIdx.x == 0) totals[d] = sh[0];
}

__global__ void __launch_bounds__(256)
radix_base_scan(const u32* __restrict__ totals, u32* base) {
  __shared__ u32 sh[256];
  int t = threadIdx.x;
  sh[t] = totals[t];
  __syncthreads();
  for (int off = 1; off < 256; off <<= 1) {
    u32 add = (t >= off) ? sh[t - off] : 0u;
    __syncthreads();
    sh[t] += add;
    __syncthreads();
  }
  base[t] = (t == 0) ? 0u : sh[t - 1];
}

__global__ void radix_digit_offsets(u32* counts, int numBlocks,
                                    const u32* __restrict__ base) {
  if (threadIdx.x != 0) return;
  int d = blockIdx.x;
  u32 run = base[d];
  size_t o = (size_t)d * numBlocks;
  for (int b = 0; b < numBlocks; ++b) {
    u32 c = counts[o + b];
    counts[o + b] = run;
    run += c;
  }
}

// one wave32 per block; stable in-batch ranking via ballot
__global__ void __launch_bounds__(32)
radix_scatter(const u64* __restrict__ src, u64* __restrict__ dst,
              int n, int shift, const u32* __restrict__ counts, int numBlocks) {
  __shared__ u32 run[RADIX];
  int b = blockIdx.x, lane = threadIdx.x;
  for (int i = lane; i < RADIX; i += 32)
    run[i] = counts[(size_t)i * numBlocks + b];
  __syncthreads();
  int start = b * TILE, end = imin(n, start + TILE);
  for (int i0 = start; i0 < end; i0 += 32) {
    int i = i0 + lane;
    bool valid = i < end;
    u64 k = valid ? src[i] : 0ull;
    u32 d = (u32)((k >> shift) & 0xFF);
    u32 validMask = __builtin_amdgcn_ballot_w32(valid);
    u32 eq = validMask;
#pragma unroll
    for (int bit = 0; bit < 8; ++bit) {
      u32 bal = __builtin_amdgcn_ballot_w32((d >> bit) & 1);
      eq &= ((d >> bit) & 1) ? bal : ~bal;
    }
    u32 rank = __popc(eq & ((1u << lane) - 1u));
    if (valid) dst[run[d] + rank] = k;
    __syncthreads();
    if (valid && rank == 0) run[d] += __popc(eq);     // group leader bumps counter
    __syncthreads();
  }
}

// ---------------------------------------------------------------
// 4) finalize kernels
// ---------------------------------------------------------------
__global__ void topk_finalize(const u64* __restrict__ keys, const int* __restrict__ batch,
                              int* node_map, float* __restrict__ out, int k) {
  int i = blockIdx.x * blockDim.x + threadIdx.x;
  if (i >= k) return;
  int idx = (int)(keys[i] & 0x1FFFF);
  node_map[idx] = i;
  out[O_PERM + i]  = (float)idx;
  out[O_BATCH + i] = (float)batch[idx];
}

__global__ void __launch_bounds__(256)
gather_scale(const u64* __restrict__ keys, const float* __restrict__ x,
             const float* __restrict__ score, float* __restrict__ out) {
  int i = blockIdx.x;                       // new row id, grid = KSEL
  int idx = (int)(keys[i] & 0x1FFFF);
  float s = score[idx];
  out[O_XNEW + (size_t)i * DIM + threadIdx.x] =
      x[(size_t)idx * DIM + threadIdx.x] * s;
}

__global__ void edge_finalize(const u64* __restrict__ keys,
                              const float* __restrict__ edge_attr,
                              const int* __restrict__ node_map,
                              float* __restrict__ out, int ne) {
  int e = blockIdx.x * blockDim.x + threadIdx.x;
  if (e >= ne) return;
  u64 k = keys[e];
  int pe = (int)(k & 0x3FFFFF);
  int r  = (int)((k >> 39) & 0x1FFFF);      // row_s
  int c  = (int)((k >> 22) & 0x1FFFF);      // col_s
  int mr = node_map[c];                     // ei_swapped[0] = col_s
  int mc = node_map[r];                     // ei_swapped[1] = row_s
  bool msk = (mr >= 0) && (mc >= 0);
  out[O_EI + e]        = (float)(msk ? mr : -1);
  out[O_EI + ne + e]   = (float)(msk ? mc : -1);
  out[O_EA + e]        = msk ? edge_attr[pe] : 0.0f;
  out[O_MASK + e]      = msk ? 1.0f : 0.0f;
}

// ---------------------------------------------------------------
// host side
// ---------------------------------------------------------------
static void run_radix_sort(u64* kA, u64* kB, int n, int passes,
                           u32* counts, u32* totals, u32* base,
                           u64** result, hipStream_t s) {
  int nb = (n + TILE - 1) / TILE;
  u64 *src = kA, *dst = kB;
  for (int p = 0; p < passes; ++p) {
    int shift = p * 8;
    radix_hist<<<nb, 256, 0, s>>>(src, n, shift, counts, nb);
    radix_digit_totals<<<256, 256, 0, s>>>(counts, nb, totals);
    radix_base_scan<<<1, 256, 0, s>>>(totals, base);
    radix_digit_offsets<<<256, 32, 0, s>>>(counts, nb, base);
    radix_scatter<<<nb, 32, 0, s>>>(src, dst, n, shift, counts, nb);
    u64* t = src; src = dst; dst = t;
  }
  *result = src;   // after swap, src holds sorted data
}

extern "C" void kernel_launch(void* const* d_in, const int* in_sizes, int n_in,
                              void* d_out, int out_size, void* d_ws, size_t ws_size,
                              hipStream_t stream) {
  const float* x          = (const float*)d_in[0];
  const int*   edge_index = (const int*)d_in[1];
  const float* edge_attr  = (const float*)d_in[2];
  const int*   batch      = (const int*)d_in[3];
  const float* p          = (const float*)d_in[4];
  const float* beta       = (const float*)d_in[5];
  float* out = (float*)d_out;

  // ---- carve workspace ----
  char* w = (char*)d_ws;
  auto carve = [&](size_t bytes) -> void* {
    void* r = (void*)w;
    w += (bytes + 255) & ~(size_t)255;
    return r;
  };
  float* score1   = (float*)carve((size_t)NNODES * 4);
  float* score2   = (float*)carve((size_t)NNODES * 4);
  float* score    = (float*)carve((size_t)NNODES * 4);
  int*   node_map = (int*)  carve((size_t)NNODES * 4);
  u64*   nkA      = (u64*)  carve((size_t)NNODES * 8);
  u64*   nkB      = (u64*)  carve((size_t)NNODES * 8);
  u64*   ekA      = (u64*)  carve((size_t)NEDGES * 8);
  u64*   ekB      = (u64*)  carve((size_t)NEDGES * 8);
  int eBlocks = (NEDGES + TILE - 1) / TILE;
  u32*   counts   = (u32*)  carve((size_t)RADIX * eBlocks * 4);
  u32*   totals   = (u32*)  carve((size_t)RADIX * 4);
  u32*   base     = (u32*)  carve((size_t)RADIX * 4);

  const int gN  = (NNODES + 255) / 256;
  const int gNE = (NEDGES + 255) / 256;

  // ---- scores ----
  int nTiles = NNODES / 16;                               // 6250, exact
  gemv_wmma<<<(nTiles + 7) / 8, 256, 0, stream>>>(x, p, score1, nTiles);
  fill_u32<<<gN, 256, 0, stream>>>((u32*)score2, 0u, NNODES);
  seg_sum<<<gNE, 256, 0, stream>>>(edge_index + NEDGES, edge_attr, score2, NEDGES);
  score_kernel<<<gN, 256, 0, stream>>>(score1, score2, beta, score, NNODES);

  // ---- top-K: sort nodes by (descending score, index) ----
  build_node_keys<<<gN, 256, 0, stream>>>(score, nkA, NNODES);
  u64* nodeSorted = nullptr;
  run_radix_sort(nkA, nkB, NNODES, 7, counts, totals, base, &nodeSorted, stream);

  fill_u32<<<gN, 256, 0, stream>>>((u32*)node_map, 0xFFFFFFFFu, NNODES);
  topk_finalize<<<(KSEL + 255) / 256, 256, 0, stream>>>(nodeSorted, batch,
                                                        node_map, out, KSEL);
  gather_scale<<<KSEL, 256, 0, stream>>>(nodeSorted, x, score, out);

  // ---- edge lexsort by (row, col, edge-id) then remap ----
  build_edge_keys<<<gNE, 256, 0, stream>>>(edge_index, ekA, NEDGES);
  u64* edgeSorted = nullptr;
  run_radix_sort(ekA, ekB, NEDGES, 7, counts, totals, base, &edgeSorted, stream);

  edge_finalize<<<gNE, 256, 0, stream>>>(edgeSorted, edge_attr, node_map,
                                         out, NEDGES);
}
